// SpectralLanguageModel_27212912788371
// MI455X (gfx1250) — compile-verified, matrix operand order checked by reading, not run on hardware
//
#include <hip/hip_runtime.h>
#include <hip/hip_bf16.h>
#include <math.h>

// ---------------------------------------------------------------------------
// Types for CDNA5 WMMA (wave32): v_wmma_f32_16x16x32_bf16
// ---------------------------------------------------------------------------
typedef __attribute__((ext_vector_type(16))) __bf16 bf16x16;
typedef __attribute__((ext_vector_type(8)))  float  floatx8;

#define TWO_PI 6.283185307179586f

// GEMM tiling
#define KC   128      // K-chunk staged in LDS
#define KCP  136      // padded row stride (272B) -> no LDS bank conflicts

// ---------------------------------------------------------------------------
// CDNA5 async copy global->LDS (ASYNCcnt) - inline asm per bridge doc
// ---------------------------------------------------------------------------
__device__ inline void async_load_b128(unsigned lds_off, const void* gptr) {
    asm volatile("global_load_async_to_lds_b128 %0, %1, off"
                 :: "v"(lds_off), "v"((unsigned long long)(size_t)gptr)
                 : "memory");
}

// ---------------------------------------------------------------------------
// A fragment (16x32 bf16) direct from global, row-major, row stride ld.
// Lane l: l&15 = row, l>>4 selects K halves; two 16B loads.
// ---------------------------------------------------------------------------
__device__ inline bf16x16 load_frag_g(const __bf16* __restrict__ base, int ld,
                                      int rc0, int k0, int lane) {
    int m  = lane & 15;
    int hi = lane >> 4;
    const __bf16* p = base + (size_t)(rc0 + m) * (size_t)ld + k0 + 8 * hi;
    bf16x16 f;
    uint4* u = reinterpret_cast<uint4*>(&f);
    u[0] = *reinterpret_cast<const uint4*>(p);
    u[1] = *reinterpret_cast<const uint4*>(p + 16);
    return f;
}

// B fragment from the LDS panel (row = N-col within panel, stride KCP)
__device__ inline bf16x16 load_frag_l(const __bf16* buf, int nrow0, int kloc,
                                      int lane) {
    int n  = lane & 15;
    int hi = lane >> 4;
    const __bf16* p = buf + (nrow0 + n) * KCP + kloc + 8 * hi;
    bf16x16 f;
    uint4* u = reinterpret_cast<uint4*>(&f);
    u[0] = *reinterpret_cast<const uint4*>(p);      // ds_load_b128
    u[1] = *reinterpret_cast<const uint4*>(p + 16);
    return f;
}

// ---------------------------------------------------------------------------
// WMMA GEMM:  C[M,N](f32) = A[M,K](bf16 row-major) * Bt[N,K](bf16)^T + bias
// EPI==0: C = acc+bias    EPI==1: C += acc+bias (residual)
// Block = 256 thr (8 waves); block tile 256(M) x 64(N); wave tile 32x64.
// B panel (64 x KC) double-buffered in LDS via async-to-LDS copies.
// Requires M%256==0, N%64==0, K%KC==0 (true for all GEMMs in this model).
// ---------------------------------------------------------------------------
template <int EPI>
__global__ __launch_bounds__(256) void gemm_bf16_kernel(
    const __bf16* __restrict__ A, const __bf16* __restrict__ Bt,
    const float* __restrict__ bias, float* __restrict__ C,
    int M, int N, int K) {
    __shared__ __bf16 bpanel[2][64 * KCP];

    int tid  = threadIdx.x;
    int lane = tid & 31;
    int w    = tid >> 5;
    int nBlk = N >> 6;
    int mBlk = blockIdx.x / nBlk;
    int nb   = blockIdx.x - mBlk * nBlk;
    int col0 = nb << 6;
    int row0 = (mBlk << 8) + (w << 5);          // wave's 32-row slice

    // ---- async issue of one B K-chunk into LDS buffer bi (4 x 16B / thread)
    auto issueB = [&](int bi, int kc) {
        unsigned base_off = (unsigned)(size_t)(&bpanel[bi][0]);
#pragma unroll
        for (int r = 0; r < 4; ++r) {
            int idx = tid + r * 256;            // 0..1023
            int rowp = idx >> 4;                // panel row (0..63)
            int c8   = idx & 15;                // 16B chunk within row
            unsigned lds_off = base_off + (unsigned)(rowp * KCP + c8 * 8) * 2u;
            const __bf16* g = Bt + (size_t)(col0 + rowp) * (size_t)K
                                 + kc * KC + c8 * 8;
            async_load_b128(lds_off, g);
        }
    };

    floatx8 acc[2][4];
#pragma unroll
    for (int i = 0; i < 2; ++i)
#pragma unroll
        for (int j = 0; j < 4; ++j) acc[i][j] = floatx8{};

    int nk = K / KC;
    issueB(0, 0);

    for (int kc = 0; kc < nk; ++kc) {
        int cur = kc & 1;
        if (kc + 1 < nk) {
            issueB(cur ^ 1, kc + 1);
            asm volatile("s_wait_asynccnt 0x4" ::: "memory");  // cur chunk done
        } else {
            asm volatile("s_wait_asynccnt 0x0" ::: "memory");
        }
        __syncthreads();

        const __bf16* buf = &bpanel[cur][0];
#pragma unroll
        for (int ks = 0; ks < KC; ks += 32) {
            int kg = kc * KC + ks;
            if (ks + 32 < KC || kc + 1 < nk) {  // prefetch next A K-slab
                __builtin_prefetch(A + (size_t)(row0 + (lane & 15)) * K + kg + 32, 0, 1);
            }
            bf16x16 a0 = load_frag_g(A, K, row0,      kg, lane);
            bf16x16 a1 = load_frag_g(A, K, row0 + 16, kg, lane);
            bf16x16 b0 = load_frag_l(buf, 0,  ks, lane);
            bf16x16 b1 = load_frag_l(buf, 16, ks, lane);
            bf16x16 b2 = load_frag_l(buf, 32, ks, lane);
            bf16x16 b3 = load_frag_l(buf, 48, ks, lane);
            acc[0][0] = __builtin_amdgcn_wmma_f32_16x16x32_bf16(false, a0, false, b0, (short)0, acc[0][0], false, false);
            acc[0][1] = __builtin_amdgcn_wmma_f32_16x16x32_bf16(false, a0, false, b1, (short)0, acc[0][1], false, false);
            acc[0][2] = __builtin_amdgcn_wmma_f32_16x16x32_bf16(false, a0, false, b2, (short)0, acc[0][2], false, false);
            acc[0][3] = __builtin_amdgcn_wmma_f32_16x16x32_bf16(false, a0, false, b3, (short)0, acc[0][3], false, false);
            acc[1][0] = __builtin_amdgcn_wmma_f32_16x16x32_bf16(false, a1, false, b0, (short)0, acc[1][0], false, false);
            acc[1][1] = __builtin_amdgcn_wmma_f32_16x16x32_bf16(false, a1, false, b1, (short)0, acc[1][1], false, false);
            acc[1][2] = __builtin_amdgcn_wmma_f32_16x16x32_bf16(false, a1, false, b2, (short)0, acc[1][2], false, false);
            acc[1][3] = __builtin_amdgcn_wmma_f32_16x16x32_bf16(false, a1, false, b3, (short)0, acc[1][3], false, false);
        }
        __syncthreads();   // all waves done reading before buffer is reloaded
    }

    // C/D layout: lane l -> N = l&15 (+subtile*16), VGPR r -> M = 8*(l>>4)+r
    int nl = lane & 15, hi = lane >> 4;
    float bv0 = bias[col0 + nl];
    float bv1 = bias[col0 + 16 + nl];
    float bv2 = bias[col0 + 32 + nl];
    float bv3 = bias[col0 + 48 + nl];
#pragma unroll
    for (int i = 0; i < 2; ++i) {
#pragma unroll
        for (int r = 0; r < 8; ++r) {
            size_t row = (size_t)(row0 + (i << 4) + (hi << 3) + r);
            float* crow = C + row * (size_t)N + col0;
            float v0 = acc[i][0][r] + bv0;
            float v1 = acc[i][1][r] + bv1;
            float v2 = acc[i][2][r] + bv2;
            float v3 = acc[i][3][r] + bv3;
            if (EPI == 1) {
                crow[nl]      += v0; crow[16 + nl] += v1;
                crow[32 + nl] += v2; crow[48 + nl] += v3;
            } else {
                crow[nl]      = v0;  crow[16 + nl] = v1;
                crow[32 + nl] = v2;  crow[48 + nl] = v3;
            }
        }
    }
}

// ---------------------------------------------------------------------------
// In-LDS iterative radix-2 FFT (DIT, data pre-bit-reversed). nthreads == N/2.
// ---------------------------------------------------------------------------
__device__ inline void fft_stages(float* re, float* im, int N, int logN,
                                  int tid, int nthreads) {
    for (int s = 1; s <= logN; ++s) {
        int hlf = 1 << (s - 1);
        int m   = 1 << s;
        __syncthreads();
        for (int bidx = tid; bidx < (N >> 1); bidx += nthreads) {
            int j   = bidx & (hlf - 1);
            int blk = bidx >> (s - 1);
            int i0  = blk * m + j;
            int i1  = i0 + hlf;
            float ang = -TWO_PI * (float)j / (float)m;
            float wi, wr;
            __sincosf(ang, &wi, &wr);
            float xr = re[i1], xi = im[i1];
            float tr = wr * xr - wi * xi;
            float ti = wr * xi + wi * xr;
            float ur = re[i0], ui = im[i0];
            re[i0] = ur + tr; im[i0] = ui + ti;
            re[i1] = ur - tr; im[i1] = ui - ti;
        }
    }
    __syncthreads();
}

// ---------------------------------------------------------------------------
// Embedding: per batch row, 2048-pt FFT of byte signal -> mag/phase (k=256)
// ---------------------------------------------------------------------------
__global__ __launch_bounds__(1024) void embed_fft_kernel(
    const int* __restrict__ byte_ids, float* __restrict__ mag,
    float* __restrict__ phase) {
    __shared__ float re[2048];
    __shared__ float im[2048];
    int b = blockIdx.x, tid = threadIdx.x;
    for (int idx = tid; idx < 2048; idx += 1024) {
        int rev = __brev((unsigned)idx) >> 21;   // 11-bit reverse
        re[rev] = (float)byte_ids[b * 2048 + idx] * (1.0f / 127.5f) - 1.0f;
        im[rev] = 0.0f;
    }
    fft_stages(re, im, 2048, 11, tid, 1024);
    if (tid < 256) {
        float xr = re[tid], xi = im[tid];
        mag[b * 256 + tid]   = sqrtf(xr * xr + xi * xi);
        phase[b * 256 + tid] = atan2f(xi, xr);
    }
}

// ---------------------------------------------------------------------------
// Features (bf16): [mag*freq_bands (256) | sin(base_phase + 2pi*t*f/T) (256)]
// ---------------------------------------------------------------------------
__global__ __launch_bounds__(256) void feats_kernel(
    const float* __restrict__ mag, const float* __restrict__ phase,
    const float* __restrict__ fb, __bf16* __restrict__ F) {
    int token = blockIdx.x;       // b*2048 + t
    int b = token >> 11;
    int t = token & 2047;
    int f = threadIdx.x;          // 0..255
    __bf16* row = F + (size_t)token * 512;
    row[f] = (__bf16)(mag[b * 256 + f] * fb[f]);
    float ph = phase[b * 256 + f] + (TWO_PI / 2048.0f) * (float)t * (float)f;
    row[256 + f] = (__bf16)__sinf(ph);
}

// ---------------------------------------------------------------------------
// Row LayerNorm (+optional exact GELU) -> bf16. One block per row, 256 thr.
// ---------------------------------------------------------------------------
__global__ __launch_bounds__(256) void ln_to_bf16_kernel(
    const float* __restrict__ X, const float* __restrict__ g,
    const float* __restrict__ bb, __bf16* __restrict__ Y, int N, int doGelu) {
    __shared__ float r1[256];
    __shared__ float r2[256];
    int row = blockIdx.x, tid = threadIdx.x;
    const float* xr = X + (size_t)row * N;
    float s = 0.f, s2 = 0.f;
    for (int i = tid; i < N; i += 256) { float v = xr[i]; s += v; s2 += v * v; }
    r1[tid] = s; r2[tid] = s2;
    __syncthreads();
    for (int o = 128; o > 0; o >>= 1) {
        if (tid < o) { r1[tid] += r1[tid + o]; r2[tid] += r2[tid + o]; }
        __syncthreads();
    }
    float mean = r1[0] / (float)N;
    float var  = r2[0] / (float)N - mean * mean;
    float inv  = rsqrtf(var + 1e-5f);
    for (int i = tid; i < N; i += 256) {
        float v = (xr[i] - mean) * inv * g[i] + bb[i];
        if (doGelu) v = 0.5f * v * (1.0f + erff(v * 0.70710678118f));
        Y[(size_t)row * N + i] = (__bf16)v;
    }
}

// Elementwise exact GELU -> bf16
__global__ void gelu_to_bf16_kernel(const float* __restrict__ X,
                                    __bf16* __restrict__ Y, size_t n) {
    size_t i = (size_t)blockIdx.x * blockDim.x + threadIdx.x;
    if (i < n) {
        float v = X[i];
        Y[i] = (__bf16)(0.5f * v * (1.0f + erff(v * 0.70710678118f)));
    }
}

// ---------------------------------------------------------------------------
// FNet mixing, stage 1: per token, LN1 fused with 512-pt FFT over D -> z
// ---------------------------------------------------------------------------
__global__ __launch_bounds__(256) void ln_fftd_kernel(
    const float* __restrict__ X, const float* __restrict__ g,
    const float* __restrict__ bb, float* __restrict__ zre,
    float* __restrict__ zim) {
    __shared__ float re[512];
    __shared__ float im[512];
    __shared__ float r1[256];
    __shared__ float r2[256];
    int token = blockIdx.x, tid = threadIdx.x;
    const float* xr = X + (size_t)token * 512;
    float v0 = xr[tid], v1 = xr[tid + 256];
    r1[tid] = v0 + v1;
    r2[tid] = v0 * v0 + v1 * v1;
    __syncthreads();
    for (int o = 128; o > 0; o >>= 1) {
        if (tid < o) { r1[tid] += r1[tid + o]; r2[tid] += r2[tid + o]; }
        __syncthreads();
    }
    float mean = r1[0] * (1.0f / 512.0f);
    float var  = r2[0] * (1.0f / 512.0f) - mean * mean;
    float inv  = rsqrtf(var + 1e-5f);
    int rv0 = __brev((unsigned)tid) >> 23;          // 9-bit reverse
    int rv1 = __brev((unsigned)(tid + 256)) >> 23;
    re[rv0] = (v0 - mean) * inv * g[tid] + bb[tid];             im[rv0] = 0.f;
    re[rv1] = (v1 - mean) * inv * g[tid + 256] + bb[tid + 256]; im[rv1] = 0.f;
    fft_stages(re, im, 512, 9, tid, 256);
    float* zr = zre + (size_t)token * 512;
    float* zi = zim + (size_t)token * 512;
    zr[tid] = re[tid];           zr[tid + 256] = re[tid + 256];
    zi[tid] = im[tid];           zi[tid + 256] = im[tid + 256];
}

// ---------------------------------------------------------------------------
// FNet mixing, stage 2: per (b,d), 2048-pt FFT over T; x += Re(result)
// ---------------------------------------------------------------------------
__global__ __launch_bounds__(1024) void fftt_mix_kernel(
    const float* __restrict__ zre, const float* __restrict__ zim,
    float* __restrict__ X) {
    __shared__ float re[2048];
    __shared__ float im[2048];
    int b = blockIdx.x >> 9;
    int d = blockIdx.x & 511;
    int tid = threadIdx.x;
    for (int idx = tid; idx < 2048; idx += 1024) {
        int rev = __brev((unsigned)idx) >> 21;
        size_t off = ((size_t)(b * 2048 + idx)) * 512 + d;
        re[rev] = zre[off];
        im[rev] = zim[off];
    }
    fft_stages(re, im, 2048, 11, tid, 1024);
    for (int idx = tid; idx < 2048; idx += 1024) {
        size_t off = ((size_t)(b * 2048 + idx)) * 512 + d;
        X[off] += re[idx];
    }
}

// ---------------------------------------------------------------------------
// Weight convert + transpose: W[K,N] f32 -> Wt[N,K] bf16 (coalesced reads)
// ---------------------------------------------------------------------------
__global__ void convert_wt_kernel(const float* __restrict__ W,
                                  __bf16* __restrict__ Wt, int K, int N) {
    int idx = blockIdx.x * blockDim.x + threadIdx.x;
    if (idx >= K * N) return;
    int k = idx / N, n = idx - k * N;
    Wt[(size_t)n * K + k] = (__bf16)W[idx];
}

// ---------------------------------------------------------------------------
// Host side
// ---------------------------------------------------------------------------
extern "C" void kernel_launch(void* const* d_in, const int* in_sizes, int n_in,
                              void* d_out, int out_size, void* d_ws, size_t ws_size,
                              hipStream_t stream) {
    (void)in_sizes; (void)n_in; (void)out_size; (void)ws_size;
    const int B = 16, T = 2048, D = 512, L = 6, H = 1024;
    const int M = B * T;                       // 32768 tokens

    const int*   byte_ids = (const int*)  d_in[0];
    const float* fb       = (const float*)d_in[1];
    const float* fp_w1    = (const float*)d_in[2];
    const float* fp_b1    = (const float*)d_in[3];
    const float* fp_ln_g  = (const float*)d_in[4];
    const float* fp_ln_b  = (const float*)d_in[5];
    const float* fp_w2    = (const float*)d_in[6];
    const float* fp_b2    = (const float*)d_in[7];
    const float* ln1_g    = (const float*)d_in[8];
    const float* ln1_b    = (const float*)d_in[9];
    const float* ln2_g    = (const float*)d_in[10];
    const float* ln2_b    = (const float*)d_in[11];
    const float* blk_w1   = (const float*)d_in[12];
    const float* blk_b1   = (const float*)d_in[13];
    const float* blk_w2   = (const float*)d_in[14];
    const float* blk_b2   = (const float*)d_in[15];
    const float* norm_g   = (const float*)d_in[16];
    const float* norm_b   = (const float*)d_in[17];
    const float* out_w    = (const float*)d_in[18];
    const float* out_b    = (const float*)d_in[19];
    float* out = (float*)d_out;

    // ---- workspace carve-up (256B aligned) ----
    char* base = (char*)d_ws;
    size_t off = 0;
    auto take = [&](size_t bytes) {
        char* p = base + off;
        off += (bytes + 255) & ~(size_t)255;
        return p;
    };
    float*  mag   = (float*) take((size_t)B * 256 * 4);
    float*  phase = (float*) take((size_t)B * 256 * 4);
    float*  x     = (float*) take((size_t)M * D * 4);       // 64 MB
    __bf16* ab    = (__bf16*)take((size_t)M * D * 2);       // 32 MB (feats / LN outs)
    float*  h     = (float*) take((size_t)M * H * 4);       // 128 MB
    __bf16* hb    = (__bf16*)take((size_t)M * H * 2);       // 64 MB
    // FFT complex buffers alias the MLP hidden buffer (never live at once)
    float*  zre = h;
    float*  zim = h + (size_t)M * D;                        // second 64 MB of h
    __bf16* fpw1t  = (__bf16*)take((size_t)H * D * 2);
    __bf16* fpw2t  = (__bf16*)take((size_t)D * H * 2);
    __bf16* blkw1t = (__bf16*)take((size_t)L * H * D * 2);
    __bf16* blkw2t = (__bf16*)take((size_t)L * D * H * 2);
    __bf16* outwt  = (__bf16*)take((size_t)256 * D * 2);

    auto gemmGrid = [](int m, int n) { return (m >> 8) * (n >> 6); };
    const int KN = D * H;   // 524288 elements per weight matrix

    // ---- weight conversion/transpose to bf16 [N,K] ----
    convert_wt_kernel<<<(KN + 255) / 256, 256, 0, stream>>>(fp_w1, fpw1t, D, H);
    convert_wt_kernel<<<(KN + 255) / 256, 256, 0, stream>>>(fp_w2, fpw2t, H, D);
    for (int l = 0; l < L; ++l) {
        convert_wt_kernel<<<(KN + 255) / 256, 256, 0, stream>>>(
            blk_w1 + (size_t)l * KN, blkw1t + (size_t)l * KN, D, H);
        convert_wt_kernel<<<(KN + 255) / 256, 256, 0, stream>>>(
            blk_w2 + (size_t)l * KN, blkw2t + (size_t)l * KN, H, D);
    }
    convert_wt_kernel<<<(D * 256 + 255) / 256, 256, 0, stream>>>(out_w, outwt, D, 256);

    // ---- spectral embedding ----
    embed_fft_kernel<<<B, 1024, 0, stream>>>(byte_ids, mag, phase);
    feats_kernel<<<M, 256, 0, stream>>>(mag, phase, fb, ab);

    // ---- freq_proj: GEMM -> LN+GELU -> GEMM ----
    gemm_bf16_kernel<0><<<gemmGrid(M, H), 256, 0, stream>>>(ab, fpw1t, fp_b1, h, M, H, D);
    ln_to_bf16_kernel<<<M, 256, 0, stream>>>(h, fp_ln_g, fp_ln_b, hb, H, 1);
    gemm_bf16_kernel<0><<<gemmGrid(M, D), 256, 0, stream>>>(hb, fpw2t, fp_b2, x, M, D, H);

    // ---- SpectralMLPBlock stack ----
    for (int l = 0; l < L; ++l) {
        // x += Re(FFT_T(FFT_D(LN1(x))))
        ln_fftd_kernel<<<M, 256, 0, stream>>>(x, ln1_g + l * D, ln1_b + l * D, zre, zim);
        fftt_mix_kernel<<<B * D, 1024, 0, stream>>>(zre, zim, x);
        // x += gelu(LN2(x) @ W1 + b1) @ W2 + b2
        ln_to_bf16_kernel<<<M, 256, 0, stream>>>(x, ln2_g + l * D, ln2_b + l * D, ab, D, 0);
        gemm_bf16_kernel<0><<<gemmGrid(M, H), 256, 0, stream>>>(
            ab, blkw1t + (size_t)l * KN, blk_b1 + l * H, h, M, H, D);
        gelu_to_bf16_kernel<<<(int)(((size_t)M * H + 255) / 256), 256, 0, stream>>>(
            h, hb, (size_t)M * H);
        gemm_bf16_kernel<1><<<gemmGrid(M, D), 256, 0, stream>>>(
            hb, blkw2t + (size_t)l * KN, blk_b2 + l * D, x, M, D, H);
    }

    // ---- final norm + output head ----
    ln_to_bf16_kernel<<<M, 256, 0, stream>>>(x, norm_g, norm_b, ab, D, 0);
    gemm_bf16_kernel<0><<<gemmGrid(M, 256), 256, 0, stream>>>(ab, outwt, out_b, out, M, 256, D);
}
